// MAETransformerModel_76244259439065
// MI455X (gfx1250) — matrix operand order repaired and checked by reading, not compiled, not run on hardware
//
#include <hip/hip_runtime.h>
#include <math.h>
#include <stdint.h>

// ---------------- model dimensions (compile-time) ----------------
static constexpr int Bb  = 32;
static constexpr int Ss  = 512;
static constexpr int INq = 19;
static constexpr int DEe = 256;
static constexpr int DDd = 128;
static constexpr int HEh = 8;
static constexpr int HDh = 4;
static constexpr int LEe = 6;
static constexpr int LDl = 2;
static constexpr int FEf = 4 * DEe;   // 1024
static constexpr int FDf = 4 * DDd;   // 512
static constexpr int BS  = Bb * Ss;   // 16384

typedef _Float16 v16h __attribute__((ext_vector_type(16)));
typedef _Float16 v4h  __attribute__((ext_vector_type(4)));
typedef float    v8f  __attribute__((ext_vector_type(8)));
typedef int      v4i  __attribute__((ext_vector_type(4)));
typedef int      v8i  __attribute__((ext_vector_type(8)));

// ---------------- WMMA fragment helpers (CDNA5 layouts) ----------------
// A operand (16x32 f16, MxK): lane m = lane&15, hl = lane>>4.
//   element e<8  -> K = hl*8 + e ;  e>=8 -> K = 16 + hl*8 + (e-8)
__device__ __forceinline__ v16h load_afrag(const _Float16* row, int hl) {
  v16h a;
#pragma unroll
  for (int e = 0; e < 8; ++e) {
    a[e]     = row[hl * 8 + e];
    a[e + 8] = row[16 + hl * 8 + e];
  }
  return a;
}
// B operand (32x16 f16, KxN): lane n = lane&15, hl = lane>>4,
//   element e<8 -> K = hl*8 + e ;  e>=8 -> K = 16 + hl*8 + (e-8)
__device__ __forceinline__ v16h load_bfrag(const _Float16* row, int hl) {
  v16h b;
#pragma unroll
  for (int e = 0; e < 8; ++e) {
    b[e]     = row[hl * 8 + e];
    b[e + 8] = row[16 + hl * 8 + e];
  }
  return b;
}
__device__ __forceinline__ v8f wmma16(v16h a, v16h b, v8f c) {
  return __builtin_amdgcn_wmma_f32_16x16x32_f16(false, a, false, b, (short)0, c,
                                                false, false);
}

// ---------------- Tensor Data Mover: 2D f16 tile global -> LDS ----------------
// Pads 4 DWORDs after every 16 DWORDs: 32-halfword rows land on a 40-halfword
// LDS pitch (matches As/Bs/Ks padding). Descriptor built in SALU; one TDM op
// per issuing wave, tracked with TENSORcnt (in-order per wave).
__device__ __forceinline__ void tdm_load_2d_f16(unsigned lds_addr, const void* gaddr,
                                                int tile_w, int tile_h,
                                                long long stride_elems) {
  unsigned long long ga = (unsigned long long)(uintptr_t)gaddr;
  v4i g0;
  g0[0] = 1;                                      // count=1, user descriptor
  g0[1] = (int)lds_addr;                          // LDS byte address
  g0[2] = (int)(unsigned)(ga & 0xffffffffu);      // global addr lo
  g0[3] = (int)(((unsigned)(ga >> 32) & 0x01ffffffu) | (2u << 30));  // hi | type=2
  unsigned dim0 = (unsigned)tile_w, dim1 = (unsigned)tile_h;
  unsigned long long s0 = (unsigned long long)stride_elems;
  v8i g1;
  g1[0] = (int)((1u << 16)      // data_size = 2 bytes
              | (1u << 20)      // pad_enable
              | (3u << 22)      // pad_interval: 16 DWORDs
              | (3u << 25));    // pad_amount: 4 DWORDs
  g1[1] = (int)((dim0 & 0xffffu) << 16);                              // tensor_dim0 lo
  g1[2] = (int)(((dim0 >> 16) & 0xffffu) | ((dim1 & 0xffffu) << 16)); // dim0 hi | dim1 lo
  g1[3] = (int)(((dim1 >> 16) & 0xffffu) | ((unsigned)tile_w << 16)); // dim1 hi | tile_dim0
  g1[4] = (int)((unsigned)tile_h & 0xffffu);                          // tile_dim1 (tile_dim2=0)
  g1[5] = (int)(unsigned)(s0 & 0xffffffffu);                          // dim0 stride lo
  g1[6] = (int)(unsigned)((s0 >> 32) & 0xffffu);                      // dim0 stride hi
  g1[7] = 0;
  asm volatile("tensor_load_to_lds %0, %1" :: "s"(g0), "s"(g1) : "memory");
}

// ---------------- pack / order / masks ----------------
__global__ void pack_kernel(const int* __restrict__ am, const int* __restrict__ pm,
                            int* __restrict__ counts, int* __restrict__ order,
                            int* __restrict__ idx, unsigned char* __restrict__ vism,
                            unsigned char* __restrict__ encm,
                            unsigned char* __restrict__ decm) {
  int b = threadIdx.x;
  if (b >= Bb) return;
  int cnt = 0;
  for (int s = 0; s < Ss; ++s) {
    int o   = b * Ss + s;
    bool a  = am[o] != 0;
    bool p  = pm[o] != 0;
    bool vv = a && !p;
    vism[o] = vv ? 1 : 0;
    decm[o] = a ? 0 : 1;
    if (vv) { order[b * Ss + cnt] = s; ++cnt; }
    idx[o] = (cnt > 0) ? (cnt - 1) : 0;
  }
  counts[b] = cnt;
  for (int j = 0; j < Ss; ++j) encm[b * Ss + j] = (j >= cnt) ? 1 : 0;
}

// ---------------- f32 -> f16 weight conversion ----------------
__global__ void __launch_bounds__(256)
cvt16_kernel(const float* __restrict__ src, _Float16* __restrict__ dst, int n) {
  int i = blockIdx.x * 256 + threadIdx.x;
  if (i < n) dst[i] = (_Float16)src[i];
}

// ---------------- gather + input projection (K=19, VALU) ----------------
__global__ void __launch_bounds__(256)
embed_kernel(const float* __restrict__ ev, const float* __restrict__ w,
             const float* __restrict__ bias, const int* __restrict__ order,
             const int* __restrict__ counts, float* __restrict__ X,
             _Float16* __restrict__ X16) {
  int bs = blockIdx.x;
  int b  = bs / Ss, j = bs % Ss;
  __shared__ float e[32];
  int  cnt    = counts[b];
  bool padded = (j >= cnt);
  if (!padded && threadIdx.x < INq) {
    int src = order[b * Ss + j];
    e[threadIdx.x] = ev[((size_t)(b * Ss + src)) * INq + threadIdx.x];
  }
  __syncthreads();
  for (int n = threadIdx.x; n < DEe; n += 256) {
    float acc = 0.f;
    if (!padded) {
      acc = bias[n];
#pragma unroll
      for (int k = 0; k < INq; ++k) acc += e[k] * w[n * INq + k];
    }
    float v = padded ? 0.f : acc;
    X[(size_t)bs * DEe + n]   = v;
    X16[(size_t)bs * DEe + n] = (_Float16)v;
  }
}

// ---------------- WMMA GEMM, f16 in / f32 accum, double-buffered TDM tiles ----
// C = act(A @ W^T + bias [+ R]) ; A:[M,K] f16, W:[N,K] f16.
// Block tile 128x64, 8 waves (4 M-sub x 2 N-sub), wave = 2x2 WMMA accumulators.
// LDS ping-pong: TDM for step k+1 is issued before waiting on step k
// (TENSORcnt is per-wave and in-order, so s_wait_tensorcnt 1 => tile k resident).
__global__ void __launch_bounds__(256)
gemm_wmma(const _Float16* __restrict__ A, const _Float16* __restrict__ W,
          const float* __restrict__ bias, const float* __restrict__ R,
          float* __restrict__ C32, _Float16* __restrict__ C16,
          int M, int N, int K, int relu) {
  __shared__ _Float16 As[2][128][40];   // 32 cols + TDM pad -> 20-bank row stride
  __shared__ _Float16 Bs[2][64][40];

  const int tid  = threadIdx.x;
  const int wv   = tid >> 5, lane = tid & 31;
  const int wm   = wv & 3,   wn   = wv >> 2;
  const int m0   = blockIdx.y * 128, n0 = blockIdx.x * 64;
  const int hl   = lane >> 4, l16 = lane & 15;
  const unsigned ldsA[2] = {(unsigned)(uintptr_t)&As[0][0][0],
                            (unsigned)(uintptr_t)&As[1][0][0]};
  const unsigned ldsB[2] = {(unsigned)(uintptr_t)&Bs[0][0][0],
                            (unsigned)(uintptr_t)&Bs[1][0][0]};

  v8f acc[2][2];
#pragma unroll
  for (int i = 0; i < 2; ++i)
#pragma unroll
    for (int j = 0; j < 2; ++j)
#pragma unroll
      for (int r = 0; r < 8; ++r) acc[i][j][r] = 0.f;

  const int nk = K >> 5;
  // prologue: stage tile 0 into buffer 0
  if (wv == 0)
    tdm_load_2d_f16(ldsA[0], A + (size_t)m0 * K, 32, 128, K);
  else if (wv == 1)
    tdm_load_2d_f16(ldsB[0], W + (size_t)n0 * K, 32, 64, K);

  for (int kt = 0; kt < nk; ++kt) {
    const int cur = kt & 1;
    if (kt + 1 < nk) {
      // issue next tile into the other buffer, then allow it to stay in flight
      if (wv == 0)
        tdm_load_2d_f16(ldsA[cur ^ 1], A + (size_t)m0 * K + (kt + 1) * 32, 32, 128, K);
      else if (wv == 1)
        tdm_load_2d_f16(ldsB[cur ^ 1], W + (size_t)n0 * K + (kt + 1) * 32, 32, 64, K);
      if (wv < 2) __builtin_amdgcn_s_wait_tensorcnt(1);
    } else {
      if (wv < 2) __builtin_amdgcn_s_wait_tensorcnt(0);
    }
    __syncthreads();

    v16h af[2], bf[2];
#pragma unroll
    for (int sm = 0; sm < 2; ++sm)
      af[sm] = load_afrag(&As[cur][wm * 32 + sm * 16 + l16][0], hl);
#pragma unroll
    for (int sn = 0; sn < 2; ++sn)
      bf[sn] = load_bfrag(&Bs[cur][wn * 32 + sn * 16 + l16][0], hl);
#pragma unroll
    for (int sm = 0; sm < 2; ++sm)
#pragma unroll
      for (int sn = 0; sn < 2; ++sn)
        acc[sm][sn] = wmma16(af[sm], bf[sn], acc[sm][sn]);
    __syncthreads();   // reads done before next iteration's TDM overwrites
  }

  // epilogue: C layout vgpr r -> rows r (lanes 0-15) / r+8 (lanes 16-31)
#pragma unroll
  for (int sm = 0; sm < 2; ++sm)
#pragma unroll
    for (int sn = 0; sn < 2; ++sn) {
      int n    = n0 + wn * 32 + sn * 16 + l16;
      float bv = bias ? bias[n] : 0.f;
#pragma unroll
      for (int r = 0; r < 8; ++r) {
        int m   = m0 + wm * 32 + sm * 16 + r + 8 * hl;
        float v = acc[sm][sn][r] + bv;
        if (R) v += R[(size_t)m * N + n];
        if (relu) v = fmaxf(v, 0.f);
        if (C32) C32[(size_t)m * N + n] = v;
        if (C16) C16[(size_t)m * N + n] = (_Float16)v;
      }
    }
}

// ---------------- flash-style WMMA attention (dh = 32, f16 qkv) ----------------
// grid: (S/64, H, B), block 128 (4 waves, 16 queries each)
__global__ void __launch_bounds__(128)
attn_wmma(const _Float16* __restrict__ qkv, _Float16* __restrict__ O,
          const unsigned char* __restrict__ kmask, int D, int H) {
  const int b = blockIdx.z, h = blockIdx.y, qt = blockIdx.x;
  const int tid = threadIdx.x, wv = tid >> 5, lane = tid & 31;
  const int hl  = lane >> 4, l16 = lane & 15;
  const int dh  = 32, D3 = 3 * D;

  __shared__ _Float16 Ks[64][40];        // [key][dh] + TDM pad
  __shared__ _Float16 Vt[32][72];        // [dh][key] + pad (transposed V)
  __shared__ _Float16 Ps[4][16][72];     // per-wave probability tile
  __shared__ unsigned char Ms[64];
  const unsigned ldsK = (unsigned)(uintptr_t)&Ks[0][0];

  const int qbase = b * Ss + qt * 64 + wv * 16;
  const float qsc = 0.17677669529663688f;   // 1/sqrt(32)

  v16h qf;
  {
    const _Float16* qp = qkv + (size_t)(qbase + l16) * D3 + h * dh;
#pragma unroll
    for (int e = 0; e < 8; ++e) {
      qf[e]     = (_Float16)((float)qp[hl * 8 + e] * qsc);
      qf[e + 8] = (_Float16)((float)qp[16 + hl * 8 + e] * qsc);
    }
  }

  v8f o0, o1;
  float mrun[8], lrun[8];
#pragma unroll
  for (int r = 0; r < 8; ++r) { o0[r] = 0.f; o1[r] = 0.f; mrun[r] = -1e30f; lrun[r] = 0.f; }

  for (int kt = 0; kt < Ss; kt += 64) {
    // K tile via TDM (wave 0); V transposed via cooperative f16 loads
    if (wv == 0) {
      tdm_load_2d_f16(ldsK, qkv + (size_t)(b * Ss + kt) * D3 + D + h * dh, 32, 64, D3);
      __builtin_amdgcn_s_wait_tensorcnt(0);
    }
#pragma unroll
    for (int i = 0; i < 4; ++i) {
      int e = (i * 128 + tid) * 4;
      int r = e >> 5, c = e & 31;
      const v4h vv = *(const v4h*)(qkv + (size_t)(b * Ss + kt + r) * D3 + 2 * D + h * dh + c);
      Vt[c + 0][r] = vv[0]; Vt[c + 1][r] = vv[1];
      Vt[c + 2][r] = vv[2]; Vt[c + 3][r] = vv[3];
    }
    if (tid < 64) Ms[tid] = kmask[b * Ss + kt + tid];
    __syncthreads();

    // scores: Q(16x32) @ K^T -> four 16x16 frags
    v8f sf[4];
#pragma unroll
    for (int sn = 0; sn < 4; ++sn) {
      v16h bf = load_bfrag(&Ks[sn * 16 + l16][0], hl);
      v8f z;
#pragma unroll
      for (int r = 0; r < 8; ++r) z[r] = 0.f;
      sf[sn] = wmma16(qf, bf, z);
    }

    // mask + online softmax (rows live in 16-lane halves)
    float tmax[8];
#pragma unroll
    for (int r = 0; r < 8; ++r) tmax[r] = -1e30f;
#pragma unroll
    for (int sn = 0; sn < 4; ++sn) {
      bool mk = Ms[sn * 16 + l16] != 0;
#pragma unroll
      for (int r = 0; r < 8; ++r) {
        float v = sf[sn][r];
        if (mk) v = -1e9f;
        sf[sn][r] = v;
        tmax[r]   = fmaxf(tmax[r], v);
      }
    }
#pragma unroll
    for (int off = 8; off >= 1; off >>= 1)
#pragma unroll
      for (int r = 0; r < 8; ++r)
        tmax[r] = fmaxf(tmax[r], __shfl_xor(tmax[r], off, 32));

    float rs[8];
#pragma unroll
    for (int r = 0; r < 8; ++r) {
      float mn = fmaxf(mrun[r], tmax[r]);
      float sc = __expf(mrun[r] - mn);
      mrun[r]  = mn;
      lrun[r] *= sc;
      o0[r]   *= sc;
      o1[r]   *= sc;
      rs[r]    = 0.f;
    }
#pragma unroll
    for (int sn = 0; sn < 4; ++sn)
#pragma unroll
      for (int r = 0; r < 8; ++r) {
        float p   = __expf(sf[sn][r] - mrun[r]);
        sf[sn][r] = p;
        rs[r]    += p;
      }
#pragma unroll
    for (int off = 8; off >= 1; off >>= 1)
#pragma unroll
      for (int r = 0; r < 8; ++r) rs[r] += __shfl_xor(rs[r], off, 32);
#pragma unroll
    for (int r = 0; r < 8; ++r) lrun[r] += rs[r];

    // stage P (C layout -> A operand layout via LDS)
#pragma unroll
    for (int sn = 0; sn < 4; ++sn)
#pragma unroll
      for (int r = 0; r < 8; ++r)
        Ps[wv][r + 8 * hl][sn * 16 + l16] = (_Float16)sf[sn][r];

    // O += P(16x64) @ V(64x32), split K into two 32-chunks
#pragma unroll
    for (int c = 0; c < 2; ++c) {
      v16h af = load_afrag(&Ps[wv][l16][c * 32], hl);
      {
        v16h bf = load_bfrag(&Vt[l16][c * 32], hl);
        o0 = wmma16(af, bf, o0);
      }
      {
        v16h bf = load_bfrag(&Vt[16 + l16][c * 32], hl);
        o1 = wmma16(af, bf, o1);
      }
    }
    __syncthreads();
  }

#pragma unroll
  for (int r = 0; r < 8; ++r) {
    int row   = qbase + r + 8 * hl;
    float inv = 1.f / lrun[r];
    O[(size_t)row * D + h * dh + l16]      = (_Float16)(o0[r] * inv);
    O[(size_t)row * D + h * dh + 16 + l16] = (_Float16)(o1[r] * inv);
  }
}

// ---------------- layernorm (row-wise, f32 in, f32 + f16 out) ----------------
__global__ void __launch_bounds__(128)
layernorm_kernel(const float* __restrict__ X, const float* __restrict__ g,
                 const float* __restrict__ bta, float* __restrict__ Y,
                 _Float16* __restrict__ Y16, int N) {
  const int row = blockIdx.x, t = threadIdx.x;
  const float* x = X + (size_t)row * N;
  __shared__ float red[128];
  float s = 0.f;
  for (int i = t; i < N; i += 128) s += x[i];
  red[t] = s; __syncthreads();
  for (int o = 64; o > 0; o >>= 1) { if (t < o) red[t] += red[t + o]; __syncthreads(); }
  float mean = red[0] / (float)N;
  __syncthreads();
  float v = 0.f;
  for (int i = t; i < N; i += 128) { float d = x[i] - mean; v += d * d; }
  red[t] = v; __syncthreads();
  for (int o = 64; o > 0; o >>= 1) { if (t < o) red[t] += red[t + o]; __syncthreads(); }
  float rstd = rsqrtf(red[0] / (float)N + 1e-5f);
  for (int i = t; i < N; i += 128) {
    float y = (x[i] - mean) * rstd * g[i] + bta[i];
    Y[(size_t)row * N + i]   = y;
    Y16[(size_t)row * N + i] = (_Float16)y;
  }
}

// ---------------- decoder input scatter (mask token) ----------------
__global__ void __launch_bounds__(128)
scatter_kernel(const _Float16* __restrict__ dvis16, const float* __restrict__ mtok,
               const unsigned char* __restrict__ vism, const int* __restrict__ idx,
               float* __restrict__ Xd, _Float16* __restrict__ Xd16) {
  int bs = blockIdx.x;
  int b  = bs / Ss;
  bool vis = vism[bs] != 0;
  int src  = b * Ss + idx[bs];
  for (int t = threadIdx.x; t < DDd; t += 128) {
    float v = vis ? (float)dvis16[(size_t)src * DDd + t] : mtok[t];
    Xd[(size_t)bs * DDd + t]   = v;
    Xd16[(size_t)bs * DDd + t] = (_Float16)v;
  }
}

// ---------------- output heads (two DD -> 1 projections) ----------------
__global__ void __launch_bounds__(128)
heads_kernel(const float* __restrict__ X, const float* __restrict__ qw,
             const float* __restrict__ qb, const float* __restrict__ dw,
             const float* __restrict__ db, float* __restrict__ out) {
  int row = blockIdx.x, t = threadIdx.x;
  float x = X[(size_t)row * DDd + t];
  __shared__ float r1[128], r2[128];
  r1[t] = x * qw[t];
  r2[t] = x * dw[t];
  __syncthreads();
  for (int o = 64; o > 0; o >>= 1) {
    if (t < o) { r1[t] += r1[t + o]; r2[t] += r2[t + o]; }
    __syncthreads();
  }
  if (t == 0) {
    out[(size_t)row * 2 + 0] = r1[0] + qb[0];
    out[(size_t)row * 2 + 1] = r2[0] + db[0];
  }
}

// ---------------- host orchestration ----------------
extern "C" void kernel_launch(void* const* d_in, const int* in_sizes, int n_in,
                              void* d_out, int out_size, void* d_ws, size_t ws_size,
                              hipStream_t stream) {
  (void)in_sizes; (void)n_in; (void)out_size; (void)ws_size;

  const float* event = (const float*)d_in[0];
  const int*   am    = (const int*)d_in[1];
  const int*   pm    = (const int*)d_in[2];
  const float* w_in  = (const float*)d_in[3];
  const float* b_in  = (const float*)d_in[4];
  const float* eqkvw = (const float*)d_in[5];
  const float* eqkvb = (const float*)d_in[6];
  const float* eoutw = (const float*)d_in[7];
  const float* eoutb = (const float*)d_in[8];
  const float* ef1w  = (const float*)d_in[9];
  const float* ef1b  = (const float*)d_in[10];
  const float* ef2w  = (const float*)d_in[11];
  const float* ef2b  = (const float*)d_in[12];
  const float* eg1   = (const float*)d_in[13];
  const float* eb1   = (const float*)d_in[14];
  const float* eg2   = (const float*)d_in[15];
  const float* eb2   = (const float*)d_in[16];
  const float* mtok  = (const float*)d_in[17];
  const float* e2dw  = (const float*)d_in[18];
  const float* dqkvw = (const float*)d_in[19];
  const float* dqkvb = (const float*)d_in[20];
  const float* doutw = (const float*)d_in[21];
  const float* doutb = (const float*)d_in[22];
  const float* df1w  = (const float*)d_in[23];
  const float* df1b  = (const float*)d_in[24];
  const float* df2w  = (const float*)d_in[25];
  const float* df2b  = (const float*)d_in[26];
  const float* dg1   = (const float*)d_in[27];
  const float* db1   = (const float*)d_in[28];
  const float* dg2   = (const float*)d_in[29];
  const float* db2   = (const float*)d_in[30];
  const float* qhw   = (const float*)d_in[31];
  const float* qhb   = (const float*)d_in[32];
  const float* dthw  = (const float*)d_in[33];
  const float* dthb  = (const float*)d_in[34];

  char*  ws  = (char*)d_ws;
  size_t off = 0;
  auto take = [&](size_t bytes) -> void* {
    void* p = ws + off;
    off = (off + bytes + 255) & ~(size_t)255;
    return p;
  };
  int*           counts = (int*)take((size_t)Bb * 4);
  int*           order  = (int*)take((size_t)BS * 4);
  int*           idx    = (int*)take((size_t)BS * 4);
  unsigned char* vism   = (unsigned char*)take(BS);
  unsigned char* encm   = (unsigned char*)take(BS);
  unsigned char* decm   = (unsigned char*)take(BS);
  float*     X    = (float*)take((size_t)BS * DEe * 4);     // f32 activations
  float*     T3   = (float*)take((size_t)BS * DEe * 4);     // pre-LN staging
  _Float16*  X16  = (_Float16*)take((size_t)BS * DEe * 2);  // f16 mirror of X
  _Float16*  T1h  = (_Float16*)take((size_t)BS * FEf * 2);  // qkv / ffn-hidden (f16)
  _Float16*  T2h  = (_Float16*)take((size_t)BS * DEe * 2);  // attn out / dec_vis (f16)

  // f16 weight mirrors
  struct WC { const float* src; _Float16* dst; int n; };
  _Float16* eqkvw16 = (_Float16*)take((size_t)LEe * 3 * DEe * DEe * 2);
  _Float16* eoutw16 = (_Float16*)take((size_t)LEe * DEe * DEe * 2);
  _Float16* ef1w16  = (_Float16*)take((size_t)LEe * FEf * DEe * 2);
  _Float16* ef2w16  = (_Float16*)take((size_t)LEe * DEe * FEf * 2);
  _Float16* e2dw16  = (_Float16*)take((size_t)DDd * DEe * 2);
  _Float16* dqkvw16 = (_Float16*)take((size_t)LDl * 3 * DDd * DDd * 2);
  _Float16* doutw16 = (_Float16*)take((size_t)LDl * DDd * DDd * 2);
  _Float16* df1w16  = (_Float16*)take((size_t)LDl * FDf * DDd * 2);
  _Float16* df2w16  = (_Float16*)take((size_t)LDl * DDd * FDf * 2);

  const WC wcs[] = {
      {eqkvw, eqkvw16, LEe * 3 * DEe * DEe}, {eoutw, eoutw16, LEe * DEe * DEe},
      {ef1w, ef1w16, LEe * FEf * DEe},       {ef2w, ef2w16, LEe * DEe * FEf},
      {e2dw, e2dw16, DDd * DEe},             {dqkvw, dqkvw16, LDl * 3 * DDd * DDd},
      {doutw, doutw16, LDl * DDd * DDd},     {df1w, df1w16, LDl * FDf * DDd},
      {df2w, df2w16, LDl * DDd * FDf}};
  for (const WC& wc : wcs)
    cvt16_kernel<<<(wc.n + 255) / 256, 256, 0, stream>>>(wc.src, wc.dst, wc.n);

  pack_kernel<<<1, Bb, 0, stream>>>(am, pm, counts, order, idx, vism, encm, decm);
  embed_kernel<<<BS, 256, 0, stream>>>(event, w_in, b_in, order, counts, X, X16);

  const dim3 blkG(256), blkA(128);
  // ---------- encoder: 6 layers, D=256, H=8 ----------
  for (int l = 0; l < LEe; ++l) {
    gemm_wmma<<<dim3(3 * DEe / 64, BS / 128), blkG, 0, stream>>>(
        X16, eqkvw16 + (size_t)l * 3 * DEe * DEe, eqkvb + (size_t)l * 3 * DEe,
        nullptr, nullptr, T1h, BS, 3 * DEe, DEe, 0);
    attn_wmma<<<dim3(Ss / 64, HEh, Bb), blkA, 0, stream>>>(T1h, T2h, encm, DEe, HEh);
    gemm_wmma<<<dim3(DEe / 64, BS / 128), blkG, 0, stream>>>(
        T2h, eoutw16 + (size_t)l * DEe * DEe, eoutb + (size_t)l * DEe,
        X, T3, nullptr, BS, DEe, DEe, 0);
    layernorm_kernel<<<BS, 128, 0, stream>>>(T3, eg1 + l * DEe, eb1 + l * DEe, X, X16, DEe);
    gemm_wmma<<<dim3(FEf / 64, BS / 128), blkG, 0, stream>>>(
        X16, ef1w16 + (size_t)l * FEf * DEe, ef1b + (size_t)l * FEf,
        nullptr, nullptr, T1h, BS, FEf, DEe, 1);
    gemm_wmma<<<dim3(DEe / 64, BS / 128), blkG, 0, stream>>>(
        T1h, ef2w16 + (size_t)l * DEe * FEf, ef2b + (size_t)l * DEe,
        X, T3, nullptr, BS, DEe, FEf, 0);
    layernorm_kernel<<<BS, 128, 0, stream>>>(T3, eg2 + l * DEe, eb2 + l * DEe, X, X16, DEe);
  }

  // ---------- encoder -> decoder bridge ----------
  gemm_wmma<<<dim3(DDd / 64, BS / 128), blkG, 0, stream>>>(
      X16, e2dw16, nullptr, nullptr, nullptr, T2h, BS, DDd, DEe, 0);
  scatter_kernel<<<BS, 128, 0, stream>>>(T2h, mtok, vism, idx, X, X16);

  // ---------- decoder: 2 layers, D=128, H=4 ----------
  for (int l = 0; l < LDl; ++l) {
    gemm_wmma<<<dim3(3 * DDd / 64, BS / 128), blkG, 0, stream>>>(
        X16, dqkvw16 + (size_t)l * 3 * DDd * DDd, dqkvb + (size_t)l * 3 * DDd,
        nullptr, nullptr, T1h, BS, 3 * DDd, DDd, 0);
    attn_wmma<<<dim3(Ss / 64, HDh, Bb), blkA, 0, stream>>>(T1h, T2h, decm, DDd, HDh);
    gemm_wmma<<<dim3(DDd / 64, BS / 128), blkG, 0, stream>>>(
        T2h, doutw16 + (size_t)l * DDd * DDd, doutb + (size_t)l * DDd,
        X, T3, nullptr, BS, DDd, DDd, 0);
    layernorm_kernel<<<BS, 128, 0, stream>>>(T3, dg1 + l * DDd, db1 + l * DDd, X, X16, DDd);
    gemm_wmma<<<dim3(FDf / 64, BS / 128), blkG, 0, stream>>>(
        X16, df1w16 + (size_t)l * FDf * DDd, df1b + (size_t)l * FDf,
        nullptr, nullptr, T1h, BS, FDf, DDd, 1);
    gemm_wmma<<<dim3(DDd / 64, BS / 128), blkG, 0, stream>>>(
        T1h, df2w16 + (size_t)l * DDd * FDf, df2b + (size_t)l * DDd,
        X, T3, nullptr, BS, DDd, FDf, 0);
    layernorm_kernel<<<BS, 128, 0, stream>>>(T3, dg2 + l * DDd, db2 + l * DDd, X, X16, DDd);
  }

  heads_kernel<<<BS, 128, 0, stream>>>(X, qhw, qhb, dthw, dthb, (float*)d_out);
}
